// Ternary_46995532152908
// MI455X (gfx1250) — matrix operand order repaired and checked by reading, not compiled, not run on hardware
//
#include <hip/hip_runtime.h>

typedef __attribute__((ext_vector_type(16))) _Float16 v16h;
typedef __attribute__((ext_vector_type(8)))  _Float16 v8h;
typedef __attribute__((ext_vector_type(4)))  _Float16 v4h;
typedef __attribute__((ext_vector_type(8)))  float    v8f;

#define THREADS 256

#if __has_builtin(__builtin_amdgcn_global_load_async_to_lds_b128)
#define HAS_ASYNC_LDS 1
typedef __attribute__((ext_vector_type(4))) int i4;
typedef __attribute__((address_space(1))) void* as1p;
typedef __attribute__((address_space(1))) i4*   g_i4p;  // global int4*
typedef __attribute__((address_space(3))) i4*   l_i4p;  // LDS int4*
#else
#define HAS_ASYNC_LDS 0
#endif

__device__ __forceinline__ void async_wait0() {
#if HAS_ASYNC_LDS
#if __has_builtin(__builtin_amdgcn_s_wait_asynccnt)
  __builtin_amdgcn_s_wait_asynccnt(0);
#else
  asm volatile("s_wait_asynccnt 0x0" ::: "memory");
#endif
#endif
}

// ---------------------------------------------------------------------------
// Pass 1a: X fp32 -> f16 (elementwise, fully coalesced)
// ---------------------------------------------------------------------------
__global__ __launch_bounds__(THREADS)
void cvt_x_f16(const float* __restrict__ src, _Float16* __restrict__ dst,
               long n4) {
  const long i = (long)blockIdx.x * blockDim.x + threadIdx.x;
  if (i < n4) {
    const float4 v = reinterpret_cast<const float4*>(src)[i];
    v4h h;
    h[0] = (_Float16)v.x; h[1] = (_Float16)v.y;
    h[2] = (_Float16)v.z; h[3] = (_Float16)v.w;
    reinterpret_cast<v4h*>(dst)[i] = h;
  }
}

// ---------------------------------------------------------------------------
// Pass 1b: W [K][N] fp32 -> WhT [N][K] f16.  Each thread owns one column n and
// a 64-deep k block: reads are lane-coalesced (128B per k), writes are 8x b128
// per thread filling a full 128B line of WhT.
// ---------------------------------------------------------------------------
__global__ __launch_bounds__(THREADS)
void cvt_w_transpose(const float* __restrict__ W, _Float16* __restrict__ WhT,
                     int N, int K) {
  const int n  = blockIdx.x * blockDim.x + threadIdx.x;
  const int kb = blockIdx.y * 64;
  _Float16 buf[64];
  #pragma unroll
  for (int k = 0; k < 64; ++k)
    buf[k] = (_Float16)W[(size_t)(kb + k) * N + n];
  v8h* dst = reinterpret_cast<v8h*>(WhT + (size_t)n * K + kb);
  #pragma unroll
  for (int j = 0; j < 8; ++j) {
    v8h h;
    #pragma unroll
    for (int l = 0; l < 8; ++l) h[l] = buf[j * 8 + l];
    dst[j] = h;
  }
}

// ---------------------------------------------------------------------------
// Pass 2: GEMM on pre-converted f16 operands.
// 128x128 block tile, BK=64 (two WMMA k-steps per barrier), 8 waves in a 4x2
// grid (32x64 per wave), double-buffered LDS, async global->LDS staging.
// ---------------------------------------------------------------------------
__global__ __launch_bounds__(THREADS)
void gemm_f16_wmma(const _Float16* __restrict__ Xh,   // [M][K]
                   const _Float16* __restrict__ WhT,  // [N][K]
                   float* __restrict__ Cout,
                   int M, int N, int K) {
  constexpr int BM = 128, BN = 128, BK = 64;
  __shared__ alignas(32) _Float16 As[2][BM * BK];   // [row][k]
  __shared__ alignas(32) _Float16 Bs[2][BN * BK];   // [col][k]

  const int tid  = threadIdx.x;
  const int lane = tid & 31;
  const int wid  = tid >> 5;       // 0..7
  const int wm   = wid & 3;        // 4 wave-rows of 32
  const int wn   = wid >> 2;       // 2 wave-cols of 64
  const int lrow = lane & 15;
  const int hsel = lane >> 4;      // lane group 0/1

  const int blockM = blockIdx.y * BM;
  const int blockN = blockIdx.x * BN;

  v8f acc[2][4];
  {
    v8f z = {};
    #pragma unroll
    for (int i = 0; i < 2; ++i)
      #pragma unroll
      for (int j = 0; j < 4; ++j) acc[i][j] = z;
  }

  // Stage one BMxBK (and BNxBK) f16 tile: 1024 16B chunks each, 4/thread.
  auto stage = [&](int buf, int k0) {
    #pragma unroll
    for (int i = 0; i < 4; ++i) {
      const int idx = tid + i * THREADS;     // 0..1023
      const int r   = idx >> 3;              // 8 chunks per 64-half row
      const int c   = idx & 7;
      const size_t ga = (size_t)(blockM + r) * K + k0 + c * 8;
      const size_t gb = (size_t)(blockN + r) * K + k0 + c * 8;
      const int    lo = r * BK + c * 8;
#if HAS_ASYNC_LDS
      __builtin_amdgcn_global_load_async_to_lds_b128(
          (g_i4p)(as1p)(void*)(Xh + ga), (l_i4p)(void*)&As[buf][lo], 0, 0);
      __builtin_amdgcn_global_load_async_to_lds_b128(
          (g_i4p)(as1p)(void*)(WhT + gb), (l_i4p)(void*)&Bs[buf][lo], 0, 0);
#else
      *reinterpret_cast<v8h*>(&As[buf][lo]) =
          *reinterpret_cast<const v8h*>(Xh + ga);
      *reinterpret_cast<v8h*>(&Bs[buf][lo]) =
          *reinterpret_cast<const v8h*>(WhT + gb);
#endif
    }
  };

  stage(0, 0);
  async_wait0();
  __syncthreads();

  const int KT = K / BK;
  int cur = 0;
  for (int kt = 0; kt < KT; ++kt) {
    const int nxt = cur ^ 1;
    if (kt + 1 < KT) stage(nxt, (kt + 1) * BK);

    #pragma unroll
    for (int ks = 0; ks < 2; ++ks) {          // two k=32 WMMA steps
      v16h aF[2], bF[4];
      #pragma unroll
      for (int fm = 0; fm < 2; ++fm) {
        // A 16x32 layout: lanes 0-15 hold K{0-7,16-23}, lanes 16-31 K{8-15,24-31}
        const _Float16* ap =
            &As[cur][(wm * 32 + fm * 16 + lrow) * BK + ks * 32 + hsel * 8];
        const v8h lo = *reinterpret_cast<const v8h*>(ap);
        const v8h hi = *reinterpret_cast<const v8h*>(ap + 16);
        aF[fm] = __builtin_shufflevector(
            lo, hi, 0, 1, 2, 3, 4, 5, 6, 7, 8, 9, 10, 11, 12, 13, 14, 15);
      }
      #pragma unroll
      for (int fn = 0; fn < 4; ++fn) {
        // B 32x16 layout: lanes 0-15 hold K{0-7,16-23}, lanes 16-31 K{8-15,24-31}
        const _Float16* bp =
            &Bs[cur][(wn * 64 + fn * 16 + lrow) * BK + ks * 32 + hsel * 8];
        const v8h blo = *reinterpret_cast<const v8h*>(bp);
        const v8h bhi = *reinterpret_cast<const v8h*>(bp + 16);
        bF[fn] = __builtin_shufflevector(
            blo, bhi, 0, 1, 2, 3, 4, 5, 6, 7, 8, 9, 10, 11, 12, 13, 14, 15);
      }

      #pragma unroll
      for (int fm = 0; fm < 2; ++fm)
        #pragma unroll
        for (int fn = 0; fn < 4; ++fn)
          acc[fm][fn] = __builtin_amdgcn_wmma_f32_16x16x32_f16(
              false, aF[fm], false, bF[fn],
              (short)0, acc[fm][fn], false, false);
    }

    async_wait0();
    __syncthreads();
    cur = nxt;
  }

  // Epilogue: VGPR j, lane L -> row = j + 8*(L>=16), col = L%16.
  #pragma unroll
  for (int fm = 0; fm < 2; ++fm) {
    #pragma unroll
    for (int fn = 0; fn < 4; ++fn) {
      const int row0 = blockM + wm * 32 + fm * 16 + hsel * 8;
      const int col  = blockN + wn * 64 + fn * 16 + lrow;
      #pragma unroll
      for (int j = 0; j < 8; ++j)
        Cout[(size_t)(row0 + j) * N + col] = acc[fm][fn][j];
    }
  }
}

// ---------------------------------------------------------------------------
// Fallback: fused convert-in-kernel GEMM (round-1 version) if ws is too small.
// ---------------------------------------------------------------------------
__global__ __launch_bounds__(THREADS)
void ternary_wmma_gemm_fused(const float* __restrict__ X,
                             const float* __restrict__ W,
                             float* __restrict__ Cout,
                             int M, int N, int K) {
  constexpr int BM = 128, BN = 128, BK = 32;
  __shared__ alignas(32) _Float16 As[BM * BK];
  __shared__ alignas(32) _Float16 BsT[BN * BK];

  const int tid  = threadIdx.x;
  const int lane = tid & 31;
  const int wid  = tid >> 5;
  const int wm   = wid & 3;
  const int wn   = wid >> 2;
  const int lrow = lane & 15;
  const int hsel = lane >> 4;

  const int blockM = blockIdx.y * BM;
  const int blockN = blockIdx.x * BN;

  v8f acc[2][4];
  {
    v8f z = {};
    #pragma unroll
    for (int i = 0; i < 2; ++i)
      #pragma unroll
      for (int j = 0; j < 4; ++j) acc[i][j] = z;
  }

  for (int k0 = 0; k0 < K; k0 += BK) {
    #pragma unroll
    for (int i = 0; i < 4; ++i) {
      const int idx = tid + i * THREADS;
      const int r   = idx >> 3;
      const int c4  = idx & 7;
      const float4 v = reinterpret_cast<const float4*>(
          X + (size_t)(blockM + r) * K + k0)[c4];
      v4h h;
      h[0] = (_Float16)v.x; h[1] = (_Float16)v.y;
      h[2] = (_Float16)v.z; h[3] = (_Float16)v.w;
      reinterpret_cast<v4h*>(As + r * BK)[c4] = h;
    }
    #pragma unroll
    for (int i = 0; i < 4; ++i) {
      const int idx = tid + i * THREADS;
      const int kk  = idx >> 5;
      const int c4  = idx & 31;
      const float4 v = reinterpret_cast<const float4*>(
          W + (size_t)(k0 + kk) * N + blockN)[c4];
      const int nb = c4 * 4;
      BsT[(nb + 0) * BK + kk] = (_Float16)v.x;
      BsT[(nb + 1) * BK + kk] = (_Float16)v.y;
      BsT[(nb + 2) * BK + kk] = (_Float16)v.z;
      BsT[(nb + 3) * BK + kk] = (_Float16)v.w;
    }
    __syncthreads();

    v16h aF[2], bF[4];
    #pragma unroll
    for (int fm = 0; fm < 2; ++fm) {
      const _Float16* ap = As + (wm * 32 + fm * 16 + lrow) * BK + hsel * 8;
      const v8h lo = *reinterpret_cast<const v8h*>(ap);
      const v8h hi = *reinterpret_cast<const v8h*>(ap + 16);
      aF[fm] = __builtin_shufflevector(
          lo, hi, 0, 1, 2, 3, 4, 5, 6, 7, 8, 9, 10, 11, 12, 13, 14, 15);
    }
    #pragma unroll
    for (int fn = 0; fn < 4; ++fn) {
      const _Float16* bp = BsT + (wn * 64 + fn * 16 + lrow) * BK + hsel * 8;
      const v8h blo = *reinterpret_cast<const v8h*>(bp);
      const v8h bhi = *reinterpret_cast<const v8h*>(bp + 16);
      bF[fn] = __builtin_shufflevector(
          blo, bhi, 0, 1, 2, 3, 4, 5, 6, 7, 8, 9, 10, 11, 12, 13, 14, 15);
    }

    #pragma unroll
    for (int fm = 0; fm < 2; ++fm)
      #pragma unroll
      for (int fn = 0; fn < 4; ++fn)
        acc[fm][fn] = __builtin_amdgcn_wmma_f32_16x16x32_f16(
            false, aF[fm], false, bF[fn],
            (short)0, acc[fm][fn], false, false);

    __syncthreads();
  }

  #pragma unroll
  for (int fm = 0; fm < 2; ++fm) {
    #pragma unroll
    for (int fn = 0; fn < 4; ++fn) {
      const int row0 = blockM + wm * 32 + fm * 16 + hsel * 8;
      const int col  = blockN + wn * 64 + fn * 16 + lrow;
      #pragma unroll
      for (int j = 0; j < 8; ++j)
        Cout[(size_t)(row0 + j) * N + col] = acc[fm][fn][j];
    }
  }
}

extern "C" void kernel_launch(void* const* d_in, const int* in_sizes, int n_in,
                              void* d_out, int out_size, void* d_ws, size_t ws_size,
                              hipStream_t stream) {
  const float* x = (const float*)d_in[0];   // [M,K] fp32
  const float* w = (const float*)d_in[1];   // [K,N] fp32 (ternary-valued)
  float* out = (float*)d_out;               // [M,N] fp32

  const int K = 2048;
  const int N = 2048;
  const int M = in_sizes[0] / K;            // 8192

  const size_t xh_bytes = (size_t)M * K * 2;
  const size_t wt_bytes = (size_t)N * K * 2;

  if (ws_size >= xh_bytes + wt_bytes) {
    _Float16* Xh  = (_Float16*)d_ws;
    _Float16* WhT = (_Float16*)((char*)d_ws + xh_bytes);

    const long n4 = (long)M * K / 4;
    cvt_x_f16<<<dim3((unsigned)((n4 + THREADS - 1) / THREADS)), THREADS, 0,
                stream>>>(x, Xh, n4);
    cvt_w_transpose<<<dim3(N / THREADS, K / 64), THREADS, 0, stream>>>(w, WhT,
                                                                       N, K);
    gemm_f16_wmma<<<dim3(N / 128, M / 128), THREADS, 0, stream>>>(Xh, WhT, out,
                                                                  M, N, K);
  } else {
    ternary_wmma_gemm_fused<<<dim3(N / 128, M / 128), THREADS, 0, stream>>>(
        x, w, out, M, N, K);
  }
}
